// DomainLearner_2963527434924
// MI455X (gfx1250) — matrix operand order repaired and checked by reading, not compiled, not run on hardware
//
#include <hip/hip_runtime.h>
#include <stdint.h>

// Problem dims (match reference)
#define B_ROWS 8192
#define N_FED  16384
#define IN_DIM 1024
#define LAT    512

// Workspace layout (requires >= 73 MB):
//   [ 0 MB,  1 MB) : ddT  bf16 [512][1024]    (domain_diff transposed)
//   [ 1 MB, 17 MB) : kv   bf16 [16384][512]
//   [17 MB, 33 MB) : kvT  bf16 [512][16384]
//   [33 MB, 65 MB) : Xb   bf16 [16384][1024]  (X_fed converted)
//   [65 MB, 73 MB) : encb bf16 [8192][512]    (enc converted)

typedef __bf16 v16bf __attribute__((ext_vector_type(16)));
typedef float  v8f   __attribute__((ext_vector_type(8)));

union Frag { v16bf v; uint32_t u[8]; uint4 q[2]; };

__device__ __forceinline__ uint32_t pack_bf16(float a, float b) {
    uint32_t ua = __builtin_bit_cast(uint32_t, a);
    uint32_t ub = __builtin_bit_cast(uint32_t, b);
    ua = (ua + 0x7FFFu + ((ua >> 16) & 1u)) >> 16;   // round-to-nearest-even
    ub = (ub + 0x7FFFu + ((ub >> 16) & 1u)) >> 16;
    return (ub << 16) | (ua & 0xFFFFu);
}
__device__ __forceinline__ uint16_t f2bf(float a) {
    uint32_t ua = __builtin_bit_cast(uint32_t, a);
    return (uint16_t)((ua + 0x7FFFu + ((ua >> 16) & 1u)) >> 16);
}

__device__ __forceinline__ v8f wmma_bf16(const Frag& a, const Frag& b, v8f c) {
    // (neg_a, A, neg_b, B, c_mod, C, reuse_a, reuse_b)
    return __builtin_amdgcn_wmma_f32_16x16x32_bf16(false, a.v, false, b.v,
                                                   (short)0, c, false, false);
}

// ---------------------------------------------------------------------------
// Kernel 0a: ddT[d][k] = bf16(domain_diff[k][d])   (512x1024)
// ---------------------------------------------------------------------------
__global__ void k0_transpose_dd(const float* __restrict__ dd,
                                uint16_t* __restrict__ ddT) {
    int id = blockIdx.x * blockDim.x + threadIdx.x;   // 0 .. 512*1024-1
    int d = id >> 10;
    int k = id & 1023;
    ddT[(size_t)d * IN_DIM + k] = f2bf(dd[(size_t)k * LAT + d]);
}

// ---------------------------------------------------------------------------
// Kernel 0b: elementwise f32 -> bf16, 8 elements per thread (b128 in/out)
// ---------------------------------------------------------------------------
__global__ void k0_cvt_bf16(const float* __restrict__ in,
                            uint16_t* __restrict__ outb) {
    size_t id = (size_t)blockIdx.x * blockDim.x + threadIdx.x;
    const float4* s = (const float4*)in + id * 2;
    float4 a = s[0], b = s[1];
    uint4 o;
    o.x = pack_bf16(a.x, a.y);
    o.y = pack_bf16(a.z, a.w);
    o.z = pack_bf16(b.x, b.y);
    o.w = pack_bf16(b.z, b.w);
    ((uint4*)outb)[id] = o;
}

// ---------------------------------------------------------------------------
// Kernel 1: kv = bf16(X_fed @ domain_diff); also kvT (transposed copy).
// One WG (4 waves) per 16-row strip; wave w owns 128 output cols (8 tiles).
// A fragment loaded once per K-step and reused across 8 WMMAs.
// ---------------------------------------------------------------------------
__global__ void k1_kv(const uint16_t* __restrict__ Xb,
                      const uint16_t* __restrict__ ddT,
                      uint16_t* __restrict__ kv,
                      uint16_t* __restrict__ kvT) {
    const int lane = threadIdx.x & 31;
    const int w    = threadIdx.x >> 5;
    const int hi   = (lane < 16) ? 0 : 1;
    const int l15  = lane & 15;
    const int tm   = blockIdx.x;            // 0 .. 1023  (N_FED/16)
    const int cw   = w * 128;               // wave's output column base

    const uint32_t* arow = (const uint32_t*)(Xb + (size_t)(tm * 16 + l15) * IN_DIM);
    const int adw = hi ? 4 : 0;             // A dword offset within K-step
    const int bdw = hi ? 4 : 0;             // B dword offset within K-step

    v8f acc[8];
#pragma unroll
    for (int t = 0; t < 8; ++t) acc[t] = {};

    for (int kk = 0; kk < IN_DIM; kk += 32) {
        const int kd = kk >> 1;             // dword index of K-step start
        Frag A;
        A.q[0] = *(const uint4*)(arow + kd + adw);
        A.q[1] = *(const uint4*)(arow + kd + adw + 8);
#pragma unroll
        for (int jt = 0; jt < 8; ++jt) {
            const uint32_t* brow =
                (const uint32_t*)(ddT + (size_t)(cw + jt * 16 + l15) * IN_DIM);
            Frag B;
            B.q[0] = *(const uint4*)(brow + kd + bdw);
            B.q[1] = *(const uint4*)(brow + kd + bdw + 8);
            acc[jt] = wmma_bf16(A, B, acc[jt]);
        }
    }

#pragma unroll
    for (int jt = 0; jt < 8; ++jt) {
        const int col = cw + jt * 16 + l15;
        // kvT: rows contiguous -> one b128 store of 8 bf16
        uint4 o;
        o.x = pack_bf16(acc[jt][0], acc[jt][1]);
        o.y = pack_bf16(acc[jt][2], acc[jt][3]);
        o.z = pack_bf16(acc[jt][4], acc[jt][5]);
        o.w = pack_bf16(acc[jt][6], acc[jt][7]);
        *(uint4*)(kvT + (size_t)col * N_FED + tm * 16 + (hi ? 8 : 0)) = o;
        // kv: strided scalar stores (row stride = LAT)
#pragma unroll
        for (int r = 0; r < 8; ++r) {
            int row = tm * 16 + r + (hi ? 8 : 0);
            kv[(size_t)row * LAT + col] = f2bf(acc[jt][r]);
        }
    }
}

// ---------------------------------------------------------------------------
// Kernel 2: fused scores + online softmax + z accumulation (flash style).
// 128 threads (4 waves) per 16 enc rows; wave w owns latent slice [128w,128w+128).
// ---------------------------------------------------------------------------
__global__ void k2_attn(const uint16_t* __restrict__ encb,
                        const uint16_t* __restrict__ kv,
                        const uint16_t* __restrict__ kvT,
                        float* __restrict__ out) {
    __shared__ float    sPart[4][16][32];   // per-wave partial score tiles
    __shared__ uint32_t pT[16][16];         // softmax probs, bf16 pairs
    __shared__ float    alphaS[16];         // per-row rescale factor
    __shared__ float    ellS[16];           // per-row running sum

    const int lane = threadIdx.x & 31;
    const int w    = threadIdx.x >> 5;
    const int hi   = (lane < 16) ? 0 : 1;
    const int l15  = lane & 15;
    const int rb   = blockIdx.x * 16;       // enc row block
    const int dw   = w * 128;               // this wave's latent slice
    const int bdw  = hi ? 4 : 0;            // B dword offset
    const int adw  = hi ? 4 : 0;            // A dword offset

    // Persistent enc A-fragments for this wave's K-slice (4 x 16x32 bf16).
    Frag encA[4];
    {
        const uint32_t* er = (const uint32_t*)(encb + (size_t)(rb + l15) * LAT);
#pragma unroll
        for (int f = 0; f < 4; ++f) {
            const int kd = (dw + f * 32) >> 1;
            encA[f].q[0] = *(const uint4*)(er + kd + adw);
            encA[f].q[1] = *(const uint4*)(er + kd + adw + 8);
        }
    }

    v8f zacc[8];
#pragma unroll
    for (int t = 0; t < 8; ++t) zacc[t] = {};

    float m_run = -1e30f, l_run = 0.0f;     // meaningful in wave 0 lanes 0-15

    for (int n0 = 0; n0 < N_FED; n0 += 32) {
        // Prefetch next n-tile (speculative; lands in near caches / L2 hit)
        __builtin_prefetch((const void*)(kv + (size_t)(n0 + 32 + l15) * LAT + dw), 0, 1);
        __builtin_prefetch((const void*)(kvT + (size_t)(dw + l15) * N_FED + n0 + 32), 0, 1);

        // ---- partial scores: enc_slice @ kv_sliceT (two 16x16 N-tiles) ----
        v8f s0 = {}, s1 = {};
#pragma unroll
        for (int f = 0; f < 4; ++f) {
            Frag b0, b1;
            const uint32_t* r0 =
                (const uint32_t*)(kv + (size_t)(n0 + l15) * LAT + dw + f * 32);
            const uint32_t* r1 =
                (const uint32_t*)(kv + (size_t)(n0 + 16 + l15) * LAT + dw + f * 32);
            b0.q[0] = *(const uint4*)(r0 + bdw);
            b0.q[1] = *(const uint4*)(r0 + bdw + 8);
            b1.q[0] = *(const uint4*)(r1 + bdw);
            b1.q[1] = *(const uint4*)(r1 + bdw + 8);
            s0 = wmma_bf16(encA[f], b0, s0);
            s1 = wmma_bf16(encA[f], b1, s1);
        }
#pragma unroll
        for (int r = 0; r < 8; ++r) {
            int row = r + (hi ? 8 : 0);
            sPart[w][row][l15]      = s0[r];
            sPart[w][row][16 + l15] = s1[r];
        }
        __syncthreads();

        // ---- online softmax (wave 0, one lane per row) ----
        if (w == 0 && lane < 16) {
            const int r = lane;
            float sv[32];
#pragma unroll
            for (int cc = 0; cc < 8; ++cc) {
                float4 a = *(const float4*)&sPart[0][r][cc * 4];
                float4 b = *(const float4*)&sPart[1][r][cc * 4];
                float4 c = *(const float4*)&sPart[2][r][cc * 4];
                float4 d = *(const float4*)&sPart[3][r][cc * 4];
                sv[cc * 4 + 0] = (a.x + b.x + c.x + d.x) * (1.0f / 512.0f);
                sv[cc * 4 + 1] = (a.y + b.y + c.y + d.y) * (1.0f / 512.0f);
                sv[cc * 4 + 2] = (a.z + b.z + c.z + d.z) * (1.0f / 512.0f);
                sv[cc * 4 + 3] = (a.w + b.w + c.w + d.w) * (1.0f / 512.0f);
            }
            float tmax = sv[0];
#pragma unroll
            for (int c = 1; c < 32; ++c) tmax = fmaxf(tmax, sv[c]);
            float m_new = fmaxf(m_run, tmax);
            float alpha = __expf(m_run - m_new);
            float lsum  = 0.0f;
#pragma unroll
            for (int c = 0; c < 32; c += 2) {
                float p0 = __expf(sv[c]     - m_new);
                float p1 = __expf(sv[c + 1] - m_new);
                lsum += p0 + p1;
                pT[r][c >> 1] = pack_bf16(p0, p1);
            }
            l_run = l_run * alpha + lsum;
            m_run = m_new;
            alphaS[r] = alpha;
            ellS[r]   = l_run;
        }
        __syncthreads();

        // ---- rescale z, then z += p @ kv_tile over this wave's 128 cols ----
        {
            float al[8];
#pragma unroll
            for (int r = 0; r < 8; ++r) al[r] = alphaS[r + (hi ? 8 : 0)];
#pragma unroll
            for (int t = 0; t < 8; ++t)
#pragma unroll
                for (int r = 0; r < 8; ++r) zacc[t][r] *= al[r];

            Frag pA;
            const int h4 = hi ? 4 : 0;
            pA.q[0] = *(const uint4*)&pT[l15][h4];        // dwords h4 .. h4+3
            pA.q[1] = *(const uint4*)&pT[l15][8 + h4];    // dwords 8+h4 .. 11+h4

#pragma unroll
            for (int jt = 0; jt < 8; ++jt) {
                Frag bz;
                const uint32_t* rz =
                    (const uint32_t*)(kvT + (size_t)(dw + jt * 16 + l15) * N_FED + n0);
                bz.q[0] = *(const uint4*)(rz + bdw);
                bz.q[1] = *(const uint4*)(rz + bdw + 8);
                zacc[jt] = wmma_bf16(pA, bz, zacc[jt]);
            }
        }
        __syncthreads();
    }

    // ---- normalize by running sum and store ----
#pragma unroll
    for (int r = 0; r < 8; ++r) {
        int row = r + (hi ? 8 : 0);
        float inv = 1.0f / ellS[row];
#pragma unroll
        for (int jt = 0; jt < 8; ++jt)
            out[(size_t)(rb + row) * LAT + dw + jt * 16 + l15] = zacc[jt][r] * inv;
    }
}

// ---------------------------------------------------------------------------
extern "C" void kernel_launch(void* const* d_in, const int* in_sizes, int n_in,
                              void* d_out, int out_size, void* d_ws, size_t ws_size,
                              hipStream_t stream) {
    const float* enc = (const float*)d_in[0];   // (8192, 512)
    const float* Xf  = (const float*)d_in[1];   // (16384, 1024)
    const float* dd  = (const float*)d_in[2];   // (1024, 512)
    float* out = (float*)d_out;                 // (8192, 512)

    char* ws = (char*)d_ws;
    uint16_t* ddT  = (uint16_t*)(ws);                        //  1 MB
    uint16_t* kv   = (uint16_t*)(ws + (size_t)( 1u << 20));  // 16 MB
    uint16_t* kvT  = (uint16_t*)(ws + (size_t)(17u << 20));  // 16 MB
    uint16_t* Xb   = (uint16_t*)(ws + (size_t)(33u << 20));  // 32 MB
    uint16_t* encb = (uint16_t*)(ws + (size_t)(65u << 20));  //  8 MB

    // Stage 0: conversions (bandwidth-bound, ~100 MB total traffic)
    k0_transpose_dd<<<(512 * 1024) / 256, 256, 0, stream>>>(dd, ddT);
    k0_cvt_bf16<<<(N_FED * IN_DIM / 8) / 256, 256, 0, stream>>>(Xf, Xb);
    k0_cvt_bf16<<<(B_ROWS * LAT / 8) / 256, 256, 0, stream>>>(enc, encb);
    // Stage 1: kv GEMM, one WG per 16-row strip of N_FED
    k1_kv<<<N_FED / 16, 128, 0, stream>>>(Xb, ddT, kv, kvT);
    // Stage 2: fused attention, 16 rows per workgroup
    k2_attn<<<B_ROWS / 16, 128, 0, stream>>>(encb, kv, kvT, out);
}
